// LocalGNOBlock_10539849744712
// MI455X (gfx1250) — compile-verified, matrix operand order checked by reading, not run on hardware
//
#include <hip/hip_runtime.h>

// ---------------------------------------------------------------------------
// LocalGNOBlock fused CDNA5 kernel (gfx1250, wave32, bf16 WMMA, fp32 accum)
//
// Key algebra: msg-MLP first layer is linear in concat([hi,hj,d]) so
//   pre(i,p) = PA[i] + PB[p] + (c[p]-c[i])*wd + b1   for p = i +/- 1..6
// where PA = h@W1[:128], PB = h@W1[128:256], wd = W1[256].
// Second layer is linear -> apply once to (sum_p silu(pre))/cnt, bias folds
// to +b2.  Update MLP: concat([h,agg])@U1 = h@U1t + agg@U1b.
// ---------------------------------------------------------------------------

typedef __bf16 v16bf __attribute__((ext_vector_type(16)));
typedef float  v8f   __attribute__((ext_vector_type(8)));

#define HD       128
#define TILE     64
#define HALO     16
#define RRR      (TILE + 2 * HALO)   // 96 compute rows (halo covers K=6)
#define NWAVES   8
#define NTHREADS 256
#define KCONN    6

// LDS layout (byte offsets), buffers aliased by stage lifetime
#define SM_XBF   0                   // bf16 h tile+halo       96*128*2 = 24576
#define SM_CC    24576               // f32 coords             96*4
#define SM_PA    25088               // f32 PA                 96*128*4 = 49152
#define SM_PB    74240               // f32 PB                 96*128*4 = 49152
#define SM_SB    123392              // bf16 S/cnt             64*128*2 = 16384
#define SM_GB    SM_PA               // bf16 agg   (PA dead)   16384
#define SM_UB    41472               // bf16 upd hidden        16384 (in PA region)
#define SM_XOUT  SM_PB               // f32 h+dh  (PB dead)    64*128*4 = 32768
#define SM_TOTAL 139776              // ~136.5 KB  (< 320 KB WGP LDS)

// Fast SiLU: v_exp + v_add + v_rcp + v_mul (avoids the ~10-op precise-div
// expansion; 1-ulp rcp is negligible next to bf16 GEMM inputs).
__device__ __forceinline__ float silu(float x) {
  return x * __builtin_amdgcn_rcpf(1.0f + __expf(-x));
}

__device__ __forceinline__ v8f wmma_bf16(v16bf a, v16bf b, v8f c) {
  // v_wmma_f32_16x16x32_bf16  (neg_a, A, neg_b, B, c_mod, C, reuse_a, reuse_b)
  return __builtin_amdgcn_wmma_f32_16x16x32_bf16(false, a, false, b, (short)0, c,
                                                 false, false);
}

// Load one 16x32 bf16 fragment (A layout) from a row-major [*,128] array.
// Works for A-tiles (row = M) and for B-tiles from pre-transposed weights
// (row = N, element = W[k][n]).  Layout per ISA 7.12.2:
//   lane 0-15  row=lane,   slots 0-7 -> K 0-7,  slots 8-15 -> K 16-23
//   lane 16-31 row=lane-16, +8 on all K
__device__ __forceinline__ v16bf load_tile16x32(const __bf16* __restrict__ base,
                                                int row0, int kbase) {
  const int lane = (int)(threadIdx.x & 31);
  const int r = row0 + (lane & 15);
  const int kl = kbase + ((lane >> 4) << 3);
  const __bf16* p = base + (size_t)r * HD + kl;
  v16bf v;
#pragma unroll
  for (int s = 0; s < 16; ++s) {
    const int k = s + ((s >> 3) << 3);   // 0..7 -> 0..7 ; 8..15 -> 16..23
    v[s] = p[k];
  }
  return v;
}

// C/D 16x16 f32 layout: lane 0-15: vgpr r -> M=r, N=lane;
//                       lane 16-31: vgpr r -> M=r+8, N=lane-16
__device__ __forceinline__ void store_tile_f32(float* __restrict__ base,
                                               int row0, int col0, v8f d) {
  const int lane = (int)(threadIdx.x & 31);
  const int n = col0 + (lane & 15);
  const int mb = row0 + ((lane >> 4) << 3);
#pragma unroll
  for (int r = 0; r < 8; ++r)
    base[(size_t)(mb + r) * HD + n] = d[r];
}

__device__ __forceinline__ void store_tile_bf16(__bf16* __restrict__ base,
                                                int row0, int col0, v8f d,
                                                const float* __restrict__ bias,
                                                bool do_silu) {
  const int lane = (int)(threadIdx.x & 31);
  const int n = col0 + (lane & 15);
  const int mb = row0 + ((lane >> 4) << 3);
  const float bv = bias ? bias[n] : 0.0f;
#pragma unroll
  for (int r = 0; r < 8; ++r) {
    float x = d[r] + bv;
    if (do_silu) x = silu(x);
    base[(size_t)(mb + r) * HD + n] = (__bf16)x;
  }
}

// ---------------------------------------------------------------------------
// Prologue: convert + transpose the six 128x128 weight blocks to bf16 in ws.
// ws[mat*16384 + n*128 + k] = W[k][n]  so B-fragments gather row-contiguous.
//   mat 0: msg_w1[0:128]   1: msg_w1[128:256]   2: msg_w2
//   mat 3: upd_w1[0:128]   4: upd_w1[128:256]   5: upd_w2
// ---------------------------------------------------------------------------
__global__ void gno_wconv(const float* __restrict__ msg_w1,
                          const float* __restrict__ msg_w2,
                          const float* __restrict__ upd_w1,
                          const float* __restrict__ upd_w2,
                          __bf16* __restrict__ ws) {
  int e = (int)(blockIdx.x * blockDim.x + threadIdx.x);   // 0..98303
  int mat = e >> 14;
  int n = (e >> 7) & 127;
  int k = e & 127;
  const float* src;
  int srow;
  switch (mat) {
    case 0: src = msg_w1; srow = k;       break;
    case 1: src = msg_w1; srow = k + 128; break;
    case 2: src = msg_w2; srow = k;       break;
    case 3: src = upd_w1; srow = k;       break;
    case 4: src = upd_w1; srow = k + 128; break;
    default: src = upd_w2; srow = k;      break;
  }
  ws[e] = (__bf16)src[(size_t)srow * HD + n];
}

// ---------------------------------------------------------------------------
// Fused main kernel: one WG per (batch, 64-node tile)
// ---------------------------------------------------------------------------
__global__ __launch_bounds__(NTHREADS)
void gno_kernel(const float* __restrict__ h, const float* __restrict__ coord,
                const float* __restrict__ msg_w1, const float* __restrict__ msg_b1,
                const float* __restrict__ msg_b2,
                const float* __restrict__ upd_b1, const float* __restrict__ upd_b2,
                const float* __restrict__ ln_g, const float* __restrict__ ln_b,
                const __bf16* __restrict__ wt, float* __restrict__ out, int N) {
  __shared__ __align__(16) char smem[SM_TOTAL];
  __bf16* Xbf = (__bf16*)(smem + SM_XBF);
  float*  CC  = (float*)(smem + SM_CC);
  float*  PA  = (float*)(smem + SM_PA);
  float*  PB  = (float*)(smem + SM_PB);
  __bf16* SB  = (__bf16*)(smem + SM_SB);
  __bf16* GB  = (__bf16*)(smem + SM_GB);
  __bf16* UB  = (__bf16*)(smem + SM_UB);
  float*  XO  = (float*)(smem + SM_XOUT);

  const int tid  = (int)threadIdx.x;
  const int wave = tid >> 5;
  const int lane = tid & 31;
  const int bidx = (int)blockIdx.y;
  const int tileBase = (int)blockIdx.x * TILE;
  const int r0 = tileBase - HALO;

  const __bf16* W1T = wt;
  const __bf16* W1B = wt + 1 * 16384;
  const __bf16* W2  = wt + 2 * 16384;
  const __bf16* U1T = wt + 3 * 16384;
  const __bf16* U1B = wt + 4 * 16384;
  const __bf16* U2  = wt + 5 * 16384;

  // ---- Stage 0: load h tile+halo -> bf16 LDS; coords -> LDS -----------------
  for (int e = tid; e < RRR * HD; e += NTHREADS) {
    int rr = e >> 7, c = e & 127;
    int g = r0 + rr;
    float v = (g >= 0 && g < N) ? h[((size_t)bidx * N + g) * HD + c] : 0.0f;
    Xbf[e] = (__bf16)v;
  }
  if (tid < RRR) {
    int g = r0 + tid;
    CC[tid] = (g >= 0 && g < N) ? coord[(size_t)bidx * N + g] : 0.0f;
  }
  __syncthreads();

  // ---- Stage 1: PA = Xbf@W1T, PB = Xbf@W1B over 96 rows (fp32 in LDS) ------
  for (int idx = wave; idx < 96; idx += NWAVES) {
    int mat = idx / 48;
    int rt  = (idx % 48) >> 3;
    int ct  = idx & 7;
    const __bf16* W = mat ? W1B : W1T;
    v8f acc = {};
#pragma unroll
    for (int kb = 0; kb < 4; ++kb) {
      v16bf a = load_tile16x32(Xbf, rt * 16, kb * 32);
      v16bf b = load_tile16x32(W, ct * 16, kb * 32);
      acc = wmma_bf16(a, b, acc);
    }
    store_tile_f32(mat ? PB : PA, rt * 16, ct * 16, acc);
  }
  __syncthreads();

  // ---- Stage 2: S = sum over 12 neighbors of silu(pre), /cnt -> bf16 -------
  {
    const int c = tid & 127;
    const float wdc = msg_w1[256 * HD + c];
    const float b1c = msg_b1[c];
    for (int j = 0; j < 32; ++j) {
      int n  = (tid >> 7) + 2 * j;       // 0..63
      int li = n + HALO;
      int gi = tileBase + n;
      float pa = PA[(size_t)li * HD + c];
      float ci = CC[li];
      float s = 0.0f;
#pragma unroll
      for (int off = 1; off <= KCONN; ++off) {
        if (gi + off < N) {
          float pre = pa + PB[(size_t)(li + off) * HD + c] +
                      (CC[li + off] - ci) * wdc + b1c;
          s += silu(pre);
        }
        if (gi - off >= 0) {
          float pre = pa + PB[(size_t)(li - off) * HD + c] +
                      (CC[li - off] - ci) * wdc + b1c;
          s += silu(pre);
        }
      }
      int cl = gi < KCONN ? gi : KCONN;
      int cr = (N - 1 - gi) < KCONN ? (N - 1 - gi) : KCONN;
      int cnt = cl + cr;
      if (cnt < 1) cnt = 1;
      float inv = __builtin_amdgcn_rcpf((float)cnt);   // v_rcp_f32, exact for
      SB[(size_t)n * HD + c] = (__bf16)(s * inv);      // small pow2-ish ints
    }
  }
  __syncthreads();

  // ---- Stage 3: agg = SB@W2 + b2 -> bf16 (aliased over PA) ------------------
  for (int idx = wave; idx < 32; idx += NWAVES) {
    int rt = idx >> 3, ct = idx & 7;
    v8f acc = {};
#pragma unroll
    for (int kb = 0; kb < 4; ++kb) {
      v16bf a = load_tile16x32(SB, rt * 16, kb * 32);
      v16bf b = load_tile16x32(W2, ct * 16, kb * 32);
      acc = wmma_bf16(a, b, acc);
    }
    store_tile_bf16(GB, rt * 16, ct * 16, acc, msg_b2, false);
  }
  __syncthreads();

  // ---- Stage 4: U = silu(h@U1T + agg@U1B + b1) -> bf16 ----------------------
  for (int idx = wave; idx < 32; idx += NWAVES) {
    int rt = idx >> 3, ct = idx & 7;
    v8f acc = {};
#pragma unroll
    for (int kb = 0; kb < 4; ++kb) {
      v16bf a = load_tile16x32(Xbf, HALO + rt * 16, kb * 32);
      v16bf b = load_tile16x32(U1T, ct * 16, kb * 32);
      acc = wmma_bf16(a, b, acc);
    }
#pragma unroll
    for (int kb = 0; kb < 4; ++kb) {
      v16bf a = load_tile16x32(GB, rt * 16, kb * 32);
      v16bf b = load_tile16x32(U1B, ct * 16, kb * 32);
      acc = wmma_bf16(a, b, acc);
    }
    store_tile_bf16(UB, rt * 16, ct * 16, acc, upd_b1, true);
  }
  __syncthreads();

  // ---- Stage 5: x = h + (U@U2 + b2)  -> f32 (aliased over PB) ---------------
  for (int idx = wave; idx < 32; idx += NWAVES) {
    int rt = idx >> 3, ct = idx & 7;
    v8f acc = {};
#pragma unroll
    for (int kb = 0; kb < 4; ++kb) {
      v16bf a = load_tile16x32(UB, rt * 16, kb * 32);
      v16bf b = load_tile16x32(U2, ct * 16, kb * 32);
      acc = wmma_bf16(a, b, acc);
    }
    const int n  = ct * 16 + (lane & 15);
    const int mb = rt * 16 + ((lane >> 4) << 3);
    const float b2v = upd_b2[n];
#pragma unroll
    for (int r = 0; r < 8; ++r) {
      int row = mb + r;
      int gi  = tileBase + row;
      float x = acc[r] + b2v + h[((size_t)bidx * N + gi) * HD + n];
      XO[(size_t)row * HD + n] = x;
    }
  }
  __syncthreads();

  // ---- Stage 6: LayerNorm per row, wave-shuffle reductions ------------------
  for (int rr = 0; rr < 8; ++rr) {
    int row = wave * 8 + rr;
    int gi  = tileBase + row;
    float v[4];
    float s0 = 0.0f, s1 = 0.0f;
#pragma unroll
    for (int j = 0; j < 4; ++j) {
      v[j] = XO[(size_t)row * HD + lane + 32 * j];
      s0 += v[j];
      s1 += v[j] * v[j];
    }
#pragma unroll
    for (int m = 16; m >= 1; m >>= 1) {
      s0 += __shfl_xor(s0, m, 32);
      s1 += __shfl_xor(s1, m, 32);
    }
    float mu  = s0 * (1.0f / 128.0f);
    float var = s1 * (1.0f / 128.0f) - mu * mu;
    float is  = rsqrtf(var + 1e-5f);
#pragma unroll
    for (int j = 0; j < 4; ++j) {
      int c = lane + 32 * j;
      out[((size_t)bidx * N + gi) * HD + c] =
          (v[j] - mu) * is * ln_g[c] + ln_b[c];
    }
  }
}

// ---------------------------------------------------------------------------
extern "C" void kernel_launch(void* const* d_in, const int* in_sizes, int n_in,
                              void* d_out, int out_size, void* d_ws, size_t ws_size,
                              hipStream_t stream) {
  const float* h      = (const float*)d_in[0];
  const float* coord  = (const float*)d_in[1];
  const float* msg_w1 = (const float*)d_in[2];
  const float* msg_b1 = (const float*)d_in[3];
  const float* msg_w2 = (const float*)d_in[4];
  const float* msg_b2 = (const float*)d_in[5];
  const float* upd_w1 = (const float*)d_in[6];
  const float* upd_b1 = (const float*)d_in[7];
  const float* upd_w2 = (const float*)d_in[8];
  const float* upd_b2 = (const float*)d_in[9];
  const float* ln_g   = (const float*)d_in[10];
  const float* ln_b   = (const float*)d_in[11];
  float* out = (float*)d_out;
  __bf16* ws = (__bf16*)d_ws;   // 6 * 128*128 bf16 = 196608 bytes

  const int B = 4;                       // fixed by reference setup
  const int N = in_sizes[1] / B;         // coord is (B, N)

  gno_wconv<<<6 * 16384 / NTHREADS, NTHREADS, 0, stream>>>(msg_w1, msg_w2,
                                                           upd_w1, upd_w2, ws);

  dim3 grid((N + TILE - 1) / TILE, B);
  gno_kernel<<<grid, NTHREADS, 0, stream>>>(h, coord, msg_w1, msg_b1, msg_b2,
                                            upd_b1, upd_b2, ln_g, ln_b, ws, out, N);
}
